// EventProcessor_8873402433975
// MI455X (gfx1250) — matrix operand order repaired and checked by reading, not compiled, not run on hardware
//
#include <hip/hip_runtime.h>
#include <hip/hip_bf16.h>
#include <stdint.h>

// ---------------------------------------------------------------------------
// Types for CDNA5 WMMA (wave32, 16x16x32 bf16 -> f32)
// ---------------------------------------------------------------------------
typedef __attribute__((ext_vector_type(16))) __bf16 v16bf;
typedef __attribute__((ext_vector_type(8)))  float  v8f;

// Problem constants (from reference): d=4096, d/2=2048, d/4=1024, K1=1027
#define DM      4096
#define DH      2048
#define DE      1024
#define K1_RAW  1027
#define K1_PAD  1056   // 33 * 32, bf16 row stride stays 16B-aligned (2112 B)

// ---------------------------------------------------------------------------
// Fragment loaders (per-lane k order chosen by this code, same for A and B; not the ISA table order)
// A (16x32, 16-bit): lane holds row M=lane&15, K in [(lane>>4)*16, +16) contiguous
//   -> two adjacent 16-byte chunks.
// B (32x16, 16-bit): lane holds col N=lane&15, K in [(lane>>4)*16, +16) contiguous
//   -> two adjacent 16-byte chunks.
// ---------------------------------------------------------------------------
__device__ __forceinline__ v16bf load_a_frag(const __bf16* p) {
    v16bf v;
    uint4* d = (uint4*)&v;
    d[0] = *(const uint4*)(p);        // K kbase..kbase+7
    d[1] = *(const uint4*)(p + 8);    // K kbase+8..kbase+15
    return v;
}
__device__ __forceinline__ v16bf load_b_frag(const __bf16* p) {
    v16bf v;
    uint4* d = (uint4*)&v;
    d[0] = *(const uint4*)(p);        // K kbase..kbase+7
    d[1] = *(const uint4*)(p + 8);    // K kbase+8..kbase+15
    return v;
}

// ---------------------------------------------------------------------------
// Epilogue for one 16x16 f32 C/D tile (8 VGPRs):
//   lane (0-15|16-31): col = n ; rows = row0 + j, row0 = m_base + (lane>>4)*8
// ---------------------------------------------------------------------------
template<bool RELU, bool MASK, bool OUT_BF16>
__device__ __forceinline__ void epilogue_tile(const v8f& c, void* __restrict__ Out,
                                              const float* __restrict__ bias,
                                              unsigned flags, int row0, int n, int ldo) {
    const float bv = bias[n];
    const float* cf = (const float*)&c;
#pragma unroll
    for (int j = 0; j < 8; ++j) {
        float v = cf[j] + bv;
        if (RELU) v = fmaxf(v, 0.0f);
        if (MASK && !((flags >> j) & 1u)) v = 0.0f;
        const size_t o = (size_t)(row0 + j) * (size_t)ldo + (size_t)n;
        if (OUT_BF16) ((__bf16*)Out)[o] = (__bf16)v;
        else          ((float*)Out)[o]  = v;
    }
}

// ---------------------------------------------------------------------------
// GEMM: Out[M,N] = act(A[M,K] * Bw[N,K]^T + bias), A/Bw bf16 row-major (K contig)
// Block: 256 thr = 8 waves; wave w -> rows [blockIdx.y*128 + w*16, +16),
// cols [blockIdx.x*64, +64) as 4 WMMA tiles. K multiple of 32. No divergence.
// ---------------------------------------------------------------------------
template<bool RELU, bool MASK, bool OUT_BF16>
__global__ __launch_bounds__(256) void k_gemm(const __bf16* __restrict__ A,
                                              const __bf16* __restrict__ Bw,
                                              const float*  __restrict__ bias,
                                              const int*    __restrict__ hasev,
                                              void*         __restrict__ Out,
                                              int K, int N) {
    const int lane = threadIdx.x & 31;
    const int wave = threadIdx.x >> 5;
    const int l15  = lane & 15;
    const int hi   = lane >> 4;
    const int m_base = blockIdx.y * 128 + wave * 16;
    const int n_base = blockIdx.x * 64;

    const __bf16* ap  = A  + (size_t)(m_base + l15) * K + hi * 16;
    const __bf16* bp0 = Bw + (size_t)(n_base +  0 + l15) * K + hi * 16;
    const __bf16* bp1 = Bw + (size_t)(n_base + 16 + l15) * K + hi * 16;
    const __bf16* bp2 = Bw + (size_t)(n_base + 32 + l15) * K + hi * 16;
    const __bf16* bp3 = Bw + (size_t)(n_base + 48 + l15) * K + hi * 16;

    v8f c0 = {}, c1 = {}, c2 = {}, c3 = {};
    for (int k0 = 0; k0 < K; k0 += 32) {
        const v16bf a  = load_a_frag(ap  + k0);
        const v16bf b0 = load_b_frag(bp0 + k0);
        const v16bf b1 = load_b_frag(bp1 + k0);
        const v16bf b2 = load_b_frag(bp2 + k0);
        const v16bf b3 = load_b_frag(bp3 + k0);
        c0 = __builtin_amdgcn_wmma_f32_16x16x32_bf16(false, a, false, b0, (short)0, c0, false, false);
        c1 = __builtin_amdgcn_wmma_f32_16x16x32_bf16(false, a, false, b1, (short)0, c1, false, false);
        c2 = __builtin_amdgcn_wmma_f32_16x16x32_bf16(false, a, false, b2, (short)0, c2, false, false);
        c3 = __builtin_amdgcn_wmma_f32_16x16x32_bf16(false, a, false, b3, (short)0, c3, false, false);
    }

    const int row0 = m_base + hi * 8;
    unsigned flags = 0xFFu;
    if (MASK) {
        flags = 0u;
#pragma unroll
        for (int j = 0; j < 8; ++j) flags |= (hasev[row0 + j] != 0 ? 1u : 0u) << j;
    }
    epilogue_tile<RELU, MASK, OUT_BF16>(c0, Out, bias, flags, row0, n_base +  0 + l15, N);
    epilogue_tile<RELU, MASK, OUT_BF16>(c1, Out, bias, flags, row0, n_base + 16 + l15, N);
    epilogue_tile<RELU, MASK, OUT_BF16>(c2, Out, bias, flags, row0, n_base + 32 + l15, N);
    epilogue_tile<RELU, MASK, OUT_BF16>(c3, Out, bias, flags, row0, n_base + 48 + l15, N);
}

// ---------------------------------------------------------------------------
// Per-batch argmax with first-index tie-break via 64-bit atomicMax keys.
// conf in [0,1) -> non-negative float bits are monotone as unsigned.
// ---------------------------------------------------------------------------
__global__ void k_zero_keys(unsigned long long* __restrict__ keys, int B) {
    int i = blockIdx.x * blockDim.x + threadIdx.x;
    if (i < B) keys[i] = 0ull;
}

__global__ void k_argmax(const float* __restrict__ conf, const int* __restrict__ bidx,
                         unsigned long long* __restrict__ keys, int E) {
    int e = blockIdx.x * blockDim.x + threadIdx.x;
    if (e >= E) return;
    unsigned cb = __float_as_uint(conf[e]);
    unsigned long long key =
        ((unsigned long long)cb << 32) | (unsigned long long)(0xFFFFFFFFu - (unsigned)e);
    atomicMax(&keys[bidx[e]], key);
}

// ---------------------------------------------------------------------------
// Weight conversion f32 -> bf16 (W1 padded K 1027 -> 1056 with zeros)
// ---------------------------------------------------------------------------
__global__ void k_cvt_w1(const float* __restrict__ W1, __bf16* __restrict__ W1b) {
    int i = blockIdx.x * blockDim.x + threadIdx.x;
    const int total = DH * K1_PAD;
    if (i >= total) return;
    int n = i / K1_PAD;
    int k = i - n * K1_PAD;
    float v = (k < K1_RAW) ? W1[(size_t)n * K1_RAW + k] : 0.0f;
    W1b[i] = (__bf16)v;
}

__global__ void k_cvt_w2(const float* __restrict__ W2, __bf16* __restrict__ W2b) {
    int i = blockIdx.x * blockDim.x + threadIdx.x;
    const int total = DM * DH;
    if (i >= total) return;
    W2b[i] = (__bf16)W2[i];
}

// ---------------------------------------------------------------------------
// Build bf16 feature rows [B, 1056]: [emb(1024) | conf | locx/640 | locy/480 | 0...]
// Empty slots -> zero row. Also writes has_event flags.
// ---------------------------------------------------------------------------
__global__ void k_feats(const unsigned long long* __restrict__ keys,
                        const int*   __restrict__ event_type,
                        const float* __restrict__ confidence,
                        const float* __restrict__ location,
                        const float* __restrict__ type_emb,
                        __bf16* __restrict__ feats,
                        int*    __restrict__ hasev,
                        int B) {
    int tid = blockIdx.x * blockDim.x + threadIdx.x;
    int total = B * K1_PAD;
    if (tid >= total) return;
    int b = tid / K1_PAD;
    int k = tid - b * K1_PAD;
    unsigned long long key = keys[b];
    int has = (key != 0ull);
    if (k == 0) hasev[b] = has;
    float v = 0.0f;
    if (has) {
        int idx = (int)(0xFFFFFFFFu - (unsigned)(key & 0xFFFFFFFFull));
        if (k < DE) {
            int et = event_type[idx];
            v = type_emb[(size_t)et * DE + k];
        } else if (k == DE) {
            v = confidence[idx];
        } else if (k == DE + 1) {
            v = location[(size_t)idx * 2 + 0] * (1.0f / 640.0f);
        } else if (k == DE + 2) {
            v = location[(size_t)idx * 2 + 1] * (1.0f / 480.0f);
        }
    }
    feats[(size_t)b * K1_PAD + k] = (__bf16)v;
}

// ---------------------------------------------------------------------------
// Launch
// ---------------------------------------------------------------------------
extern "C" void kernel_launch(void* const* d_in, const int* in_sizes, int n_in,
                              void* d_out, int out_size, void* d_ws, size_t ws_size,
                              hipStream_t stream) {
    const int*   event_type = (const int*)  d_in[0];
    const float* confidence = (const float*)d_in[1];
    const float* location   = (const float*)d_in[2];
    const int*   batch_idx  = (const int*)  d_in[3];
    // d_in[4] = batch_size (device scalar; B derived from out_size instead)
    const float* type_emb   = (const float*)d_in[5];
    const float* W1         = (const float*)d_in[6];
    const float* b1         = (const float*)d_in[7];
    const float* W2         = (const float*)d_in[8];
    const float* b2         = (const float*)d_in[9];

    const int E = in_sizes[1];
    const int B = out_size / DM;   // 16384

    // Workspace layout (all offsets 256-B aligned)
    char* ws = (char*)d_ws;
    unsigned long long* keys  = (unsigned long long*)(ws);                       // B*8     = 128 KB
    int*                hasev = (int*)   (ws + 131072);                          // B*4     = 64 KB
    __bf16*             W1b   = (__bf16*)(ws + 196608);                          // 2048*1056*2 ≈ 4.3 MB
    __bf16*             W2b   = (__bf16*)(ws + 4521984);                         // 4096*2048*2 = 16.8 MB
    __bf16*             feats = (__bf16*)(ws + 21299200);                        // B*1056*2 ≈ 34.6 MB
    __bf16*             h     = (__bf16*)(ws + 55902208);                        // B*2048*2 = 67 MB
    (void)ws_size; (void)n_in; (void)batch_idx;

    const int T = 256;

    // 1) reset argmax keys
    k_zero_keys<<<(B + T - 1) / T, T, 0, stream>>>(keys, B);
    // 2) per-batch argmax (max conf, tie -> first index)
    k_argmax<<<(E + T - 1) / T, T, 0, stream>>>(confidence, batch_idx, keys, E);
    // 3) weights -> bf16 (L2-resident afterwards)
    k_cvt_w1<<<(DH * K1_PAD + T - 1) / T, T, 0, stream>>>(W1, W1b);
    k_cvt_w2<<<(DM * DH + T - 1) / T, T, 0, stream>>>(W2, W2b);
    // 4) gather features
    k_feats<<<(B * K1_PAD + T - 1) / T, T, 0, stream>>>(keys, event_type, confidence,
                                                        location, type_emb, feats, hasev, B);
    // 5) GEMM1: h = relu(feats @ W1^T + b1)   [B,1056] x [2048,1056]^T -> bf16 [B,2048]
    {
        dim3 grid(DH / 64, B / 128);
        k_gemm<true, false, true><<<grid, T, 0, stream>>>(feats, W1b, b1, hasev,
                                                          (void*)h, K1_PAD, DH);
    }
    // 6) GEMM2: out = mask(h @ W2^T + b2)     [B,2048] x [4096,2048]^T -> f32 [B,4096]
    {
        dim3 grid(DM / 64, B / 128);
        k_gemm<false, true, false><<<grid, T, 0, stream>>>(h, W2b, b2, hasev,
                                                           d_out, DH, DM);
    }
}
